// Attention_88321707475088
// MI455X (gfx1250) — matrix operand order repaired and checked by reading, not compiled, not run on hardware
//
#include <hip/hip_runtime.h>
#include <hip/hip_bf16.h>
#include <math.h>

#define SEQ 2048
#define DIM 4096
#define NH  32
#define HD  128
#define NKV 8

typedef __attribute__((ext_vector_type(16))) __bf16        bf16x16;
typedef __attribute__((ext_vector_type(8)))  float         f32x8;
typedef __attribute__((ext_vector_type(4)))  float         f32x4;
typedef __attribute__((ext_vector_type(4)))  unsigned int  u32x4;
typedef __attribute__((ext_vector_type(2)))  unsigned int  u32x2;

union FragU { bf16x16 v; u32x4 q[2]; };
union PackU { __bf16 h[4]; u32x2 u; };

__device__ __forceinline__ f32x8 zero8() {
  f32x8 z = {0.f,0.f,0.f,0.f,0.f,0.f,0.f,0.f};
  return z;
}

// A fragment (16 rows x 32 K, bf16), source row-major [row][K-contig].
// ISA 7.12.2 layout: lane m=l&15; halves 0..7 <-> K = hi*8+p ; halves 8..15 <-> K = 16+hi*8+p
__device__ __forceinline__ bf16x16 frag_a(const __bf16* p, int ld, int lane) {
  int m = lane & 15, hi = lane >> 4;
  const __bf16* s = p + (size_t)m * ld + hi * 8;
  FragU f;
  f.q[0] = *(const u32x4*)(s);
  f.q[1] = *(const u32x4*)(s + 16);
  return f.v;
}

// B fragment (32 K x 16 cols, bf16), source N-major [n][K-contig].
// ISA 7.12.2 layout: lane n=l&15; halves 0..7 <-> K = hi*8+p ; halves 8..15 <-> K = 16+hi*8+p
__device__ __forceinline__ bf16x16 frag_b(const __bf16* p, int ld, int lane) {
  int n = lane & 15, hi = lane >> 4;
  const __bf16* s = p + (size_t)n * ld + hi * 8;
  FragU f;
  f.q[0] = *(const u32x4*)(s);
  f.q[1] = *(const u32x4*)(s + 16);
  return f.v;
}

__device__ __forceinline__ f32x8 wmma_bf16(bf16x16 a, bf16x16 b, f32x8 c) {
  return __builtin_amdgcn_wmma_f32_16x16x32_bf16(false, a, false, b, (short)0, c, false, false);
}

// Async copy: global -> LDS, 16B per lane, 4 rows (32 apart) per call. ASYNCcnt.
__device__ __forceinline__ void async_tile(unsigned lds, const __bf16* g, int ldg) {
  #pragma unroll
  for (int i = 0; i < 4; ++i) {
    unsigned l = lds + i * (32 * 72 * 2);
    const __bf16* p = g + (size_t)(i * 32) * ldg;
    asm volatile("global_load_async_to_lds_b128 %0, %1, off"
                 :: "v"(l), "v"(p) : "memory");
  }
}

// ---------------------------------------------------------------------------
// Elementwise fp32 -> bf16 (x)
// ---------------------------------------------------------------------------
__global__ __launch_bounds__(256) void cvt_bf16_kernel(const float* __restrict__ in,
                                                       __bf16* __restrict__ out, int n4) {
  int i = blockIdx.x * 256 + threadIdx.x;
  if (i >= n4) return;
  f32x4 v = *(const f32x4*)(in + (size_t)i * 4);
  PackU p;
  p.h[0] = (__bf16)v.x; p.h[1] = (__bf16)v.y;
  p.h[2] = (__bf16)v.z; p.h[3] = (__bf16)v.w;
  *(u32x2*)(out + (size_t)i * 4) = p.u;
}

// ---------------------------------------------------------------------------
// Transpose + convert: w[K][N] fp32 -> wT[N][K] bf16  (also used for V)
// ---------------------------------------------------------------------------
__global__ __launch_bounds__(256) void transpose_cvt_kernel(const float* __restrict__ w,
                                                            __bf16* __restrict__ wT,
                                                            int K, int N) {
  __shared__ float tile[32][33];
  int n0 = blockIdx.x * 32, k0 = blockIdx.y * 32;
  int tx = threadIdx.x & 31, ty = threadIdx.x >> 5;   // 32 x 8
  #pragma unroll
  for (int i = 0; i < 4; ++i)
    tile[ty + i * 8][tx] = w[(size_t)(k0 + ty + i * 8) * N + n0 + tx];
  __syncthreads();
  #pragma unroll
  for (int i = 0; i < 4; ++i)
    wT[(size_t)(n0 + ty + i * 8) * K + k0 + tx] = (__bf16)tile[tx][ty + i * 8];
}

// ---------------------------------------------------------------------------
// RoPE in fp32 + optional score-scale, then cast to bf16. in/out: [SEQ][H*128]
// ---------------------------------------------------------------------------
__global__ __launch_bounds__(256) void rope_cvt_kernel(const float* __restrict__ in,
                                                       __bf16* __restrict__ out,
                                                       int H, float scale) {
  int tid = blockIdx.x * 256 + threadIdx.x;
  int total = SEQ * H * 64;
  if (tid >= total) return;
  int i = tid & 63;
  int h = (tid >> 6) % H;
  int s = tid / (64 * H);
  // inv_freq = 10000^(-i/64) ; ln(10000)/64 = 0.1439115681...
  float inv = __expf(-0.14391156816f * (float)i);
  float ang = (float)s * inv;
  float sn, cs;
  __sincosf(ang, &sn, &cs);
  size_t base = ((size_t)s * H + h) * 128 + 2 * i;
  float x0 = in[base], x1 = in[base + 1];
  out[base]     = (__bf16)((x0 * cs - x1 * sn) * scale);
  out[base + 1] = (__bf16)((x0 * sn + x1 * cs) * scale);
}

// ---------------------------------------------------------------------------
// bf16 GEMM: C[M][N] fp32 = A[M][K] * BT[N][K]^T ; block tile 128x128, Ktile 64.
// Double-buffered LDS filled by global_load_async_to_lds_b128 (ASYNCcnt),
// overlapping the next tile's DMA with the current tile's WMMAs.
// ---------------------------------------------------------------------------
__global__ __launch_bounds__(256) void gemm_bf16_kernel(const __bf16* __restrict__ A,
                                                        const __bf16* __restrict__ BT,
                                                        float* __restrict__ C,
                                                        int M, int N, int K) {
  __shared__ __align__(16) __bf16 As[2][128 * 72];   // padded: 72 halves/row
  __shared__ __align__(16) __bf16 Bs[2][128 * 72];
  const int BUF_BYTES = 128 * 72 * 2;                // 18432
  int tid  = threadIdx.x;
  int lane = tid & 31, w = tid >> 5;
  int wm = (w >> 1) * 32;   // 4 wave-rows  (2 m-subtiles each)
  int wn = (w & 1)  * 64;   // 2 wave-cols  (4 n-subtiles each)
  int lr = tid >> 3, lc = (tid & 7) * 8;

  const __bf16* Ag = A  + (size_t)(blockIdx.y * 128 + lr) * K + lc;
  const __bf16* Bg = BT + (size_t)(blockIdx.x * 128 + lr) * K + lc;
  unsigned ldsA = (unsigned)(size_t)&As[0][lr * 72 + lc];
  unsigned ldsB = (unsigned)(size_t)&Bs[0][lr * 72 + lc];

  f32x8 acc[2][4];
  #pragma unroll
  for (int i = 0; i < 2; ++i)
    #pragma unroll
    for (int j = 0; j < 4; ++j) acc[i][j] = zero8();

  // prologue: tile 0 -> buffer 0 (8 async ops per thread: 4 A + 4 B)
  async_tile(ldsA, Ag, K);
  async_tile(ldsB, Bg, K);

  int nt = K / 64;
  for (int t = 0; t < nt; ++t) {
    int buf = t & 1;
    if (t + 1 < nt) {
      int nb = buf ^ 1;
      async_tile(ldsA + nb * BUF_BYTES, Ag + (size_t)(t + 1) * 64, K);
      async_tile(ldsB + nb * BUF_BYTES, Bg + (size_t)(t + 1) * 64, K);
      // in-order completion: <=8 outstanding means tile t is fully in LDS
      asm volatile("s_wait_asynccnt 0x8" ::: "memory");
    } else {
      asm volatile("s_wait_asynccnt 0x0" ::: "memory");
    }
    __syncthreads();   // all waves' async data for tile t visible

    #pragma unroll
    for (int ks = 0; ks < 2; ++ks) {
      bf16x16 af[2], bfr[4];
      #pragma unroll
      for (int i = 0; i < 2; ++i) af[i]  = frag_a(&As[buf][(wm + i * 16) * 72 + ks * 32], 72, lane);
      #pragma unroll
      for (int j = 0; j < 4; ++j) bfr[j] = frag_b(&Bs[buf][(wn + j * 16) * 72 + ks * 32], 72, lane);
      #pragma unroll
      for (int i = 0; i < 2; ++i)
        #pragma unroll
        for (int j = 0; j < 4; ++j)
          acc[i][j] = wmma_bf16(af[i], bfr[j], acc[i][j]);
    }
    __syncthreads();   // protect buffer before next iteration's async overwrite
  }

  int m = lane & 15, hi = lane >> 4;
  #pragma unroll
  for (int i = 0; i < 2; ++i)
    #pragma unroll
    for (int j = 0; j < 4; ++j)
      #pragma unroll
      for (int r = 0; r < 8; ++r) {
        int row = blockIdx.y * 128 + wm + i * 16 + hi * 8 + r;
        int col = blockIdx.x * 128 + wn + j * 16 + m;
        C[(size_t)row * N + col] = acc[i][j][r];
      }
}

// ---------------------------------------------------------------------------
// Flash attention: grid (SEQ/64, NH), 4 waves, each wave = 16 q-rows x D=128.
// qb [SEQ][NH*HD] bf16 (pre-scaled by 1/sqrt(128)), kb [SEQ][NKV*HD] bf16,
// vt [NKV*HD][SEQ] bf16 (transposed V), ao [SEQ][NH*HD] bf16.
// ---------------------------------------------------------------------------
__global__ __launch_bounds__(128) void attn_kernel(const __bf16* __restrict__ qb,
                                                   const __bf16* __restrict__ kb,
                                                   const __bf16* __restrict__ vt,
                                                   __bf16* __restrict__ ao) {
  __shared__ __align__(16) __bf16 pscr[4][16 * 40];  // per-wave P scratch, padded rows
  int lane = threadIdx.x & 31, w = threadIdx.x >> 5;
  int h = blockIdx.y, kvh = h >> 2;
  int q0 = blockIdx.x * 64 + w * 16;
  int m = lane & 15, hi = lane >> 4;

  // Q tile 16x128 resident in registers as 4 A-fragments
  bf16x16 qf[4];
  const __bf16* qbase = qb + (size_t)q0 * (NH * HD) + h * HD;
  #pragma unroll
  for (int ks = 0; ks < 4; ++ks) qf[ks] = frag_a(qbase + ks * 32, NH * HD, lane);

  f32x8 o[8];
  #pragma unroll
  for (int d = 0; d < 8; ++d) o[d] = zero8();
  float mrow[8], lrow[8];
  #pragma unroll
  for (int r = 0; r < 8; ++r) { mrow[r] = -1e30f; lrow[r] = 0.f; }

  for (int kv0 = 0; kv0 < SEQ; kv0 += 32) {
    // S = Q * K^T  (contraction d is contiguous in kb rows -> B frags contiguous)
    f32x8 sc[2];
    #pragma unroll
    for (int ns = 0; ns < 2; ++ns) {
      f32x8 a = zero8();
      const __bf16* kbase = kb + (size_t)(kv0 + ns * 16) * (NKV * HD) + kvh * HD;
      #pragma unroll
      for (int ks = 0; ks < 4; ++ks)
        a = wmma_bf16(qf[ks], frag_b(kbase + ks * 32, NKV * HD, lane), a);
      sc[ns] = a;
    }

    // online softmax over the 16x32 score tile (C layout: row = r+hi*8, col = m)
    __bf16* ps = pscr[w];
    #pragma unroll
    for (int r = 0; r < 8; ++r) {
      float mx = fmaxf(sc[0][r], sc[1][r]);
      #pragma unroll
      for (int off = 8; off >= 1; off >>= 1) mx = fmaxf(mx, __shfl_xor(mx, off, 32));
      float mnew  = fmaxf(mrow[r], mx);
      float alpha = __expf(mrow[r] - mnew);
      float p0 = __expf(sc[0][r] - mnew);
      float p1 = __expf(sc[1][r] - mnew);
      float rs = p0 + p1;
      #pragma unroll
      for (int off = 8; off >= 1; off >>= 1) rs += __shfl_xor(rs, off, 32);
      lrow[r] = lrow[r] * alpha + rs;
      mrow[r] = mnew;
      #pragma unroll
      for (int d = 0; d < 8; ++d) o[d][r] = o[d][r] * alpha;
      ps[(r + hi * 8) * 40 + m]      = (__bf16)p0;   // C-layout -> row-major LDS
      ps[(r + hi * 8) * 40 + 16 + m] = (__bf16)p1;
    }

    // re-read P as an A fragment (per-wave LDS is in-order; compiler waits DScnt)
    bf16x16 pf = frag_a(ps, 40, lane);

    // O += P * V   (vt rows = head-dim, contiguous along seq -> B frags contiguous)
    #pragma unroll
    for (int d = 0; d < 8; ++d) {
      const __bf16* vbase = vt + (size_t)(kvh * HD + d * 16) * SEQ + kv0;
      o[d] = wmma_bf16(pf, frag_b(vbase, SEQ, lane), o[d]);
    }
  }

  #pragma unroll
  for (int d = 0; d < 8; ++d)
    #pragma unroll
    for (int r = 0; r < 8; ++r) {
      float val = o[d][r] / lrow[r];
      ao[(size_t)(q0 + hi * 8 + r) * (NH * HD) + h * HD + d * 16 + m] = (__bf16)val;
    }
}

// ---------------------------------------------------------------------------
extern "C" void kernel_launch(void* const* d_in, const int* in_sizes, int n_in,
                              void* d_out, int out_size, void* d_ws, size_t ws_size,
                              hipStream_t stream) {
  (void)in_sizes; (void)n_in; (void)out_size; (void)ws_size;
  const float* x  = (const float*)d_in[0];
  const float* wq = (const float*)d_in[1];
  const float* wk = (const float*)d_in[2];
  const float* wv = (const float*)d_in[3];
  const float* wo = (const float*)d_in[4];
  float* out = (float*)d_out;

  char* ws = (char*)d_ws;
  size_t off = 0;
  auto take = [&](size_t bytes) {
    char* p = ws + off;
    off += (bytes + 255) & ~(size_t)255;
    return p;
  };

  const int KVD = NKV * HD;  // 1024
  __bf16* xb  = (__bf16*)take((size_t)SEQ * DIM * 2);
  __bf16* wqT = (__bf16*)take((size_t)DIM * DIM * 2);
  __bf16* wkT = (__bf16*)take((size_t)KVD * DIM * 2);
  __bf16* wvT = (__bf16*)take((size_t)KVD * DIM * 2);
  __bf16* woT = (__bf16*)take((size_t)DIM * DIM * 2);
  float*  qfp = (float*) take((size_t)SEQ * DIM * 4);
  float*  kfp = (float*) take((size_t)SEQ * KVD * 4);
  float*  vfp = (float*) take((size_t)SEQ * KVD * 4);
  __bf16* qbb = (__bf16*)take((size_t)SEQ * DIM * 2);
  __bf16* kbb = (__bf16*)take((size_t)SEQ * KVD * 2);
  __bf16* vt  = (__bf16*)take((size_t)KVD * SEQ * 2);
  __bf16* ao  = (__bf16*)qfp;  // alias: qfp dead after RoPE

  // 1) activations to bf16
  cvt_bf16_kernel<<<(SEQ * DIM / 4 + 255) / 256, 256, 0, stream>>>(x, xb, SEQ * DIM / 4);

  // 2) weights: convert + transpose to [N][K] bf16
  transpose_cvt_kernel<<<dim3(DIM / 32, DIM / 32), 256, 0, stream>>>(wq, wqT, DIM, DIM);
  transpose_cvt_kernel<<<dim3(KVD / 32, DIM / 32), 256, 0, stream>>>(wk, wkT, DIM, KVD);
  transpose_cvt_kernel<<<dim3(KVD / 32, DIM / 32), 256, 0, stream>>>(wv, wvT, DIM, KVD);
  transpose_cvt_kernel<<<dim3(DIM / 32, DIM / 32), 256, 0, stream>>>(wo, woT, DIM, DIM);

  // 3) projections (bf16 WMMA, fp32 out)
  gemm_bf16_kernel<<<dim3(DIM / 128, SEQ / 128), 256, 0, stream>>>(xb, wqT, qfp, SEQ, DIM, DIM);
  gemm_bf16_kernel<<<dim3(KVD / 128, SEQ / 128), 256, 0, stream>>>(xb, wkT, kfp, SEQ, KVD, DIM);
  gemm_bf16_kernel<<<dim3(KVD / 128, SEQ / 128), 256, 0, stream>>>(xb, wvT, vfp, SEQ, KVD, DIM);

  // 4) RoPE (+ 1/sqrt(HD) folded into Q) -> bf16
  rope_cvt_kernel<<<(SEQ * NH  * 64 + 255) / 256, 256, 0, stream>>>(qfp, qbb, NH, 0.08838834764831845f);
  rope_cvt_kernel<<<(SEQ * NKV * 64 + 255) / 256, 256, 0, stream>>>(kfp, kbb, NKV, 1.0f);

  // 5) V transpose: [SEQ][KVD] fp32 -> [KVD][SEQ] bf16
  transpose_cvt_kernel<<<dim3(KVD / 32, SEQ / 32), 256, 0, stream>>>(vfp, vt, SEQ, KVD);

  // 6) flash attention
  attn_kernel<<<dim3(SEQ / 64, NH), 128, 0, stream>>>(qbb, kbb, vt, ao);

  // 7) output projection -> fp32 d_out
  gemm_bf16_kernel<<<dim3(DIM / 128, SEQ / 128), 256, 0, stream>>>(ao, woT, out, SEQ, DIM, DIM);
}
